// DAMIC_88235808129614
// MI455X (gfx1250) — matrix operand order repaired and checked by reading, not compiled, not run on hardware
//
#include <hip/hip_runtime.h>
#include <hip/hip_bf16.h>

// ---------------------------------------------------------------------------
// Problem constants (from reference)
// ---------------------------------------------------------------------------
#define B_    64
#define T_    50
#define L_    64
#define LP_   72          // padded rows per utterance (zero pad for window shift)
#define EMB_  300
#define EP_   320         // E padded to multiple of 32
#define NF_   256
#define HID_  1024
#define G4_   4096        // 4*HID
#define OUT_  32
#define FEAT_ 768         // 3*NF
#define BT_   (B_ * T_)

typedef __attribute__((ext_vector_type(16))) __bf16 v16bf;
typedef __attribute__((ext_vector_type(8)))  float  v8f;

union Frag16 { v16bf v; uint4 q[2]; };

__device__ __forceinline__ unsigned short f2bf(float x) {
  unsigned int u = __float_as_uint(x);
  u += 0x7FFFu + ((u >> 16) & 1u);   // round-to-nearest-even
  return (unsigned short)(u >> 16);
}

__device__ __forceinline__ v16bf load_frag_packed(const unsigned short* p) {
  Frag16 f;
  f.q[0] = *(const uint4*)(p);
  f.q[1] = *(const uint4*)(p + 8);
  return f.v;
}

// A fragment from a row-major bf16 matrix: lane (l&15) = row, chunks at
// (l>>4)*8 and (l>>4)*8+16 along K (per 16-bit 16x32 A layout).
__device__ __forceinline__ v16bf load_a_frag(const unsigned short* rowbase_k0) {
  Frag16 f;
  f.q[0] = *(const uint4*)(rowbase_k0);
  f.q[1] = *(const uint4*)(rowbase_k0 + 16);
  return f.v;
}

// ---------------------------------------------------------------------------
// Weight packing: row-major W[N][K] fp32 -> bf16 fragments.
// Fragment (nt,kt) holds 32 lanes x 16 elems: lane l -> column n = nt*16+(l&15),
// K block kt*32 + (l>>4)*8 + (i&7) + (i>>3)*16  (B-operand striping for wmma 16x16x32 bf16).
// ---------------------------------------------------------------------------
__global__ void pack_dense_kernel(const float* __restrict__ W, int N, int K,
                                  unsigned short* __restrict__ out) {
  size_t idx = (size_t)blockIdx.x * blockDim.x + threadIdx.x;
  size_t total = (size_t)N * K;
  if (idx >= total) return;
  int ksteps = K / 32;
  int f = (int)(idx >> 9), r = (int)(idx & 511);
  int lane = r >> 4, i = r & 15;
  int nt = f / ksteps, kt = f - nt * ksteps;
  int n = nt * 16 + (lane & 15);
  int k = kt * 32 + (lane >> 4) * 8 + (i & 7) + (i >> 3) * 16;
  out[idx] = f2bf(W[(size_t)n * K + k]);
}

// Conv weights w[NF][EMB][fs] -> B[k][n] with k = j*EP + e (zero pad e>=300)
__global__ void pack_conv_kernel(const float* __restrict__ W, int fs,
                                 unsigned short* __restrict__ out) {
  int K = fs * EP_;
  size_t idx = (size_t)blockIdx.x * blockDim.x + threadIdx.x;
  size_t total = (size_t)NF_ * K;
  if (idx >= total) return;
  int ksteps = K / 32;
  int f = (int)(idx >> 9), r = (int)(idx & 511);
  int lane = r >> 4, i = r & 15;
  int nt = f / ksteps, kt = f - nt * ksteps;
  int n = nt * 16 + (lane & 15);
  int k = kt * 32 + (lane >> 4) * 8 + (i & 7) + (i >> 3) * 16;
  int j = k / EP_, e = k - j * EP_;
  float v = (e < EMB_) ? W[((size_t)n * EMB_ + e) * fs + j] : 0.0f;
  out[idx] = f2bf(v);
}

// ---------------------------------------------------------------------------
// Fused embedding-gather + conv1d(WMMA) + bias + ReLU + max-over-time.
// One workgroup (8 waves) per (utterance, filter-size).
// ---------------------------------------------------------------------------
__global__ __launch_bounds__(256) void conv_pool_kernel(
    const int* __restrict__ dialogue, const float* __restrict__ emb,
    const unsigned short* __restrict__ wpack, const float* __restrict__ bias,
    unsigned short* __restrict__ feats, int fs, int Lout, int fsoff) {
  __shared__ unsigned short sX[LP_ * EP_];   // 45 KB bf16, rows 64..71 zero

  const int bt  = blockIdx.x;
  const int tid = threadIdx.x;
  const int* tok = dialogue + bt * L_;

  for (int idx = tid; idx < L_ * EP_; idx += 256) {
    int row = idx / EP_, col = idx - row * EP_;
    float v = (col < EMB_) ? emb[(size_t)tok[row] * EMB_ + col] : 0.0f;
    sX[idx] = f2bf(v);
  }
  for (int idx = tid; idx < (LP_ - L_) * EP_; idx += 256)
    sX[L_ * EP_ + idx] = 0;
  __syncthreads();

  const int wave = tid >> 5, lane = tid & 31;
  const int mrow = lane & 15, hs = lane >> 4;
  const int ksteps = fs * (EP_ / 32);

  v8f acc[4][2];
  #pragma unroll
  for (int mt = 0; mt < 4; ++mt) { acc[mt][0] = {}; acc[mt][1] = {}; }

  for (int kt = 0; kt < ksteps; ++kt) {
    int j  = kt / (EP_ / 32);
    int ke = kt - j * (EP_ / 32);
    const unsigned short* bp0 =
        wpack + ((size_t)((wave * 2 + 0) * ksteps + kt)) * 512 + lane * 16;
    const unsigned short* bp1 =
        wpack + ((size_t)((wave * 2 + 1) * ksteps + kt)) * 512 + lane * 16;
    __builtin_prefetch(bp0 + 512, 0, 1);
    v16bf b0 = load_frag_packed(bp0);
    v16bf b1 = load_frag_packed(bp1);
    int colbase = ke * 32 + hs * 8;
    #pragma unroll
    for (int mt = 0; mt < 4; ++mt) {
      int row = mt * 16 + mrow + j;                 // shifted window row
      Frag16 a;
      a.q[0] = *(const uint4*)(&sX[row * EP_ + colbase]);
      a.q[1] = *(const uint4*)(&sX[row * EP_ + colbase + 16]);
      acc[mt][0] = __builtin_amdgcn_wmma_f32_16x16x32_bf16(
          false, a.v, false, b0, (short)0, acc[mt][0], false, false);
      acc[mt][1] = __builtin_amdgcn_wmma_f32_16x16x32_bf16(
          false, a.v, false, b1, (short)0, acc[mt][1], false, false);
    }
  }

  // Epilogue: bias, mask invalid positions, max over time, ReLU.
  const int ncol = lane & 15;
  #pragma unroll
  for (int nn = 0; nn < 2; ++nn) {
    int ng = wave * 2 + nn;
    float bv = bias[ng * 16 + ncol];
    float m = -3.4e38f;
    #pragma unroll
    for (int mt = 0; mt < 4; ++mt) {
      #pragma unroll
      for (int r = 0; r < 8; ++r) {
        int p = mt * 16 + r + 8 * hs;               // output time position
        float v = acc[mt][nn][r] + bv;
        m = fmaxf(m, (p < Lout) ? v : -3.4e38f);
      }
    }
    m = fmaxf(m, __shfl_xor(m, 16, 32));            // merge M halves
    m = fmaxf(m, 0.0f);                             // ReLU
    if (lane < 16)
      feats[(size_t)bt * FEAT_ + fsoff + ng * 16 + lane] = f2bf(m);
  }
}

// ---------------------------------------------------------------------------
// LSTM gate GEMM: gates[64][4096] = A1[64][K1] @ B1 + A2[64][K2] @ B2
// grid = (32 ngroups, 4 mtiles), 8 waves; wave -> one 16x16 n-tile.
// ---------------------------------------------------------------------------
__global__ __launch_bounds__(256) void lstm_gemm_kernel(
    const unsigned short* __restrict__ a1, int ks1, int stride1,
    const unsigned short* __restrict__ bp1,
    const unsigned short* __restrict__ a2, int ks2, int stride2,
    const unsigned short* __restrict__ bp2,
    float* __restrict__ gates) {
  const int wave = threadIdx.x >> 5, lane = threadIdx.x & 31;
  const int nt = blockIdx.x * 8 + wave;
  const int mt = blockIdx.y;
  const int mrow = lane & 15, hs = lane >> 4;

  v8f acc = {};
  const unsigned short* arow1 = a1 + (size_t)(mt * 16 + mrow) * stride1 + hs * 8;
  for (int kt = 0; kt < ks1; ++kt) {
    v16bf a = load_a_frag(arow1 + kt * 32);
    v16bf b = load_frag_packed(bp1 + ((size_t)(nt * ks1 + kt)) * 512 + lane * 16);
    acc = __builtin_amdgcn_wmma_f32_16x16x32_bf16(false, a, false, b, (short)0,
                                                  acc, false, false);
  }
  const unsigned short* arow2 = a2 + (size_t)(mt * 16 + mrow) * stride2 + hs * 8;
  for (int kt = 0; kt < ks2; ++kt) {
    v16bf a = load_a_frag(arow2 + kt * 32);
    v16bf b = load_frag_packed(bp2 + ((size_t)(nt * ks2 + kt)) * 512 + lane * 16);
    acc = __builtin_amdgcn_wmma_f32_16x16x32_bf16(false, a, false, b, (short)0,
                                                  acc, false, false);
  }
  #pragma unroll
  for (int r = 0; r < 8; ++r) {
    int m = mt * 16 + r + 8 * hs;
    gates[(size_t)m * G4_ + nt * 16 + (lane & 15)] = acc[r];
  }
}

// ---------------------------------------------------------------------------
// LSTM cell elementwise: c,h from raw gates (PyTorch i,f,g,o order).
// ---------------------------------------------------------------------------
__global__ void lstm_cell_kernel(const float* __restrict__ gates,
                                 const float* __restrict__ bvec,
                                 float* __restrict__ c,
                                 unsigned short* __restrict__ hb) {
  int idx = blockIdx.x * blockDim.x + threadIdx.x;
  if (idx >= B_ * HID_) return;
  int m = idx / HID_, k = idx - m * HID_;
  const float* g = gates + (size_t)m * G4_;
  float gi = g[k]             + bvec[k];
  float gf = g[HID_ + k]      + bvec[HID_ + k];
  float gg = g[2 * HID_ + k]  + bvec[2 * HID_ + k];
  float go = g[3 * HID_ + k]  + bvec[3 * HID_ + k];
  float si = 1.0f / (1.0f + expf(-gi));
  float sf = 1.0f / (1.0f + expf(-gf));
  float so = 1.0f / (1.0f + expf(-go));
  float c2 = sf * c[idx] + si * tanhf(gg);
  float h2 = so * tanhf(c2);
  c[idx]  = c2;
  hb[idx] = f2bf(h2);
}

// ---------------------------------------------------------------------------
// Output head: pred = sigmoid([feat_t | h1] @ h2o_w.T + b). 1 WG, 8 waves.
// Writes f32 to d_out and bf16 to prevpred (layer-0 input for next step).
// ---------------------------------------------------------------------------
__global__ __launch_bounds__(256) void pred_gemm_kernel(
    const unsigned short* __restrict__ featsb, int t,
    const unsigned short* __restrict__ h1b,
    const unsigned short* __restrict__ bph2o,
    const float* __restrict__ h2o_b,
    float* __restrict__ out, unsigned short* __restrict__ prevpred) {
  const int wave = threadIdx.x >> 5, lane = threadIdx.x & 31;
  const int mt = wave >> 1, nt = wave & 1;
  const int mrow = lane & 15, hs = lane >> 4;
  const int KS = (FEAT_ + HID_) / 32;     // 56

  v8f acc = {};
  const unsigned short* arow1 =
      featsb + ((size_t)(mt * 16 + mrow) * T_ + t) * FEAT_ + hs * 8;
  for (int kt = 0; kt < FEAT_ / 32; ++kt) {
    v16bf a = load_a_frag(arow1 + kt * 32);
    v16bf b = load_frag_packed(bph2o + ((size_t)(nt * KS + kt)) * 512 + lane * 16);
    acc = __builtin_amdgcn_wmma_f32_16x16x32_bf16(false, a, false, b, (short)0,
                                                  acc, false, false);
  }
  const unsigned short* arow2 = h1b + (size_t)(mt * 16 + mrow) * HID_ + hs * 8;
  for (int kt = 0; kt < HID_ / 32; ++kt) {
    v16bf a = load_a_frag(arow2 + kt * 32);
    v16bf b = load_frag_packed(
        bph2o + ((size_t)(nt * KS + FEAT_ / 32 + kt)) * 512 + lane * 16);
    acc = __builtin_amdgcn_wmma_f32_16x16x32_bf16(false, a, false, b, (short)0,
                                                  acc, false, false);
  }
  #pragma unroll
  for (int r = 0; r < 8; ++r) {
    int m = mt * 16 + r + 8 * hs;                   // batch index
    int n = nt * 16 + (lane & 15);                  // output channel
    float v = acc[r] + h2o_b[n];
    v = 1.0f / (1.0f + expf(-v));
    out[(size_t)m * T_ * OUT_ + t * OUT_ + n] = v;
    prevpred[m * OUT_ + n] = f2bf(v);
  }
}

// ---------------------------------------------------------------------------
// Host driver
// ---------------------------------------------------------------------------
extern "C" void kernel_launch(void* const* d_in, const int* in_sizes, int n_in,
                              void* d_out, int out_size, void* d_ws, size_t ws_size,
                              hipStream_t stream) {
  (void)in_sizes; (void)n_in; (void)out_size; (void)ws_size;
  const int*   dialogue = (const int*)  d_in[0];
  const float* emb      = (const float*)d_in[1];
  const float* cw3      = (const float*)d_in[2];
  const float* cb3      = (const float*)d_in[3];
  const float* cw4      = (const float*)d_in[4];
  const float* cb4      = (const float*)d_in[5];
  const float* cw5      = (const float*)d_in[6];
  const float* cb5      = (const float*)d_in[7];
  const float* wih0     = (const float*)d_in[8];
  const float* whh0     = (const float*)d_in[9];
  const float* b0       = (const float*)d_in[10];
  const float* wih1     = (const float*)d_in[11];
  const float* whh1     = (const float*)d_in[12];
  const float* b1       = (const float*)d_in[13];
  const float* h2o_w    = (const float*)d_in[14];
  const float* h2o_b    = (const float*)d_in[15];
  float* out = (float*)d_out;

  char* ws = (char*)d_ws;
  size_t off = 0;
  auto alloc = [&](size_t bytes) -> void* {
    void* p = ws + off;
    off = (off + bytes + 255) & ~(size_t)255;
    return p;
  };

  unsigned short* wp3   = (unsigned short*)alloc((size_t)NF_ * 3 * EP_ * 2);
  unsigned short* wp4   = (unsigned short*)alloc((size_t)NF_ * 4 * EP_ * 2);
  unsigned short* wp5   = (unsigned short*)alloc((size_t)NF_ * 5 * EP_ * 2);
  unsigned short* wpih0 = (unsigned short*)alloc((size_t)G4_ * 32 * 2);
  unsigned short* wphh0 = (unsigned short*)alloc((size_t)G4_ * HID_ * 2);
  unsigned short* wpih1 = (unsigned short*)alloc((size_t)G4_ * HID_ * 2);
  unsigned short* wphh1 = (unsigned short*)alloc((size_t)G4_ * HID_ * 2);
  unsigned short* wph2o = (unsigned short*)alloc((size_t)OUT_ * (FEAT_ + HID_) * 2);
  unsigned short* featsb = (unsigned short*)alloc((size_t)BT_ * FEAT_ * 2);
  float* gates0 = (float*)alloc((size_t)B_ * G4_ * 4);
  float* gates1 = (float*)alloc((size_t)B_ * G4_ * 4);
  float* c0     = (float*)alloc((size_t)B_ * HID_ * 4);
  float* c1     = (float*)alloc((size_t)B_ * HID_ * 4);
  unsigned short* h0b   = (unsigned short*)alloc((size_t)B_ * HID_ * 2);
  unsigned short* h1b   = (unsigned short*)alloc((size_t)B_ * HID_ * 2);
  unsigned short* pprev = (unsigned short*)alloc((size_t)B_ * OUT_ * 2);

  // Zero recurrent state each call (deterministic, graph-capture safe).
  hipMemsetAsync(c0,    0, (size_t)B_ * HID_ * 4, stream);
  hipMemsetAsync(c1,    0, (size_t)B_ * HID_ * 4, stream);
  hipMemsetAsync(h0b,   0, (size_t)B_ * HID_ * 2, stream);
  hipMemsetAsync(h1b,   0, (size_t)B_ * HID_ * 2, stream);
  hipMemsetAsync(pprev, 0, (size_t)B_ * OUT_ * 2, stream);

  const int thr = 256;
  auto blocks = [&](size_t n) { return (unsigned)((n + thr - 1) / thr); };

  pack_conv_kernel<<<blocks((size_t)NF_ * 3 * EP_), thr, 0, stream>>>(cw3, 3, wp3);
  pack_conv_kernel<<<blocks((size_t)NF_ * 4 * EP_), thr, 0, stream>>>(cw4, 4, wp4);
  pack_conv_kernel<<<blocks((size_t)NF_ * 5 * EP_), thr, 0, stream>>>(cw5, 5, wp5);
  pack_dense_kernel<<<blocks((size_t)G4_ * 32),   thr, 0, stream>>>(wih0, G4_, 32,   wpih0);
  pack_dense_kernel<<<blocks((size_t)G4_ * HID_), thr, 0, stream>>>(whh0, G4_, HID_, wphh0);
  pack_dense_kernel<<<blocks((size_t)G4_ * HID_), thr, 0, stream>>>(wih1, G4_, HID_, wpih1);
  pack_dense_kernel<<<blocks((size_t)G4_ * HID_), thr, 0, stream>>>(whh1, G4_, HID_, wphh1);
  pack_dense_kernel<<<blocks((size_t)OUT_ * (FEAT_ + HID_)), thr, 0, stream>>>(
      h2o_w, OUT_, FEAT_ + HID_, wph2o);

  conv_pool_kernel<<<BT_, 256, 0, stream>>>(dialogue, emb, wp3, cb3, featsb, 3, L_ - 2, 0);
  conv_pool_kernel<<<BT_, 256, 0, stream>>>(dialogue, emb, wp4, cb4, featsb, 4, L_ - 3, NF_);
  conv_pool_kernel<<<BT_, 256, 0, stream>>>(dialogue, emb, wp5, cb5, featsb, 5, L_ - 4, 2 * NF_);

  for (int t = 0; t < T_; ++t) {
    lstm_gemm_kernel<<<dim3(32, 4), 256, 0, stream>>>(
        pprev, 1, OUT_, wpih0, h0b, HID_ / 32, HID_, wphh0, gates0);
    lstm_cell_kernel<<<blocks((size_t)B_ * HID_), thr, 0, stream>>>(gates0, b0, c0, h0b);
    lstm_gemm_kernel<<<dim3(32, 4), 256, 0, stream>>>(
        h0b, HID_ / 32, HID_, wpih1, h1b, HID_ / 32, HID_, wphh1, gates1);
    lstm_cell_kernel<<<blocks((size_t)B_ * HID_), thr, 0, stream>>>(gates1, b1, c1, h1b);
    pred_gemm_kernel<<<1, 256, 0, stream>>>(featsb, t, h1b, wph2o, h2o_b, out, pprev);
  }
}
